// LSQWindowMSA_84885733638920
// MI455X (gfx1250) — compile-verified, matrix operand order checked by reading, not run on hardware
//
#include <hip/hip_runtime.h>

// LSQ Window-MSA, fully fused per-window kernel for gfx1250 (wave32, WMMA).
// One workgroup (256 threads = 8 wave32) per window b in [0, 4096).
// Pipeline entirely in LDS: x,w -> f16 staging -> QKV GEMM -> QK^T -> softmax
// (+rel-pos bias + shift-window mask) -> P@V -> proj GEMM -> d_out.
//
// Round-2 changes (same WMMA mix, less overhead around it):
//  - phase 1 is nt-outer: wqkv B-tiles loaded once per nt, reused for 4 mt
//  - qkv scatter store is branchless (computed base+stride, single ds_store)
//  - no integer division in tile decomposition
//  - phase 4 is mt-outer: attn-out A-tiles loaded once, reused for 4 nt

#define NTOK 49
#define NPAD 64
#define CDIM 128
#define HEADS 4
#define DHEAD 32

typedef __attribute__((ext_vector_type(16))) _Float16 v16h;
typedef __attribute__((ext_vector_type(4)))  _Float16 v4h;
typedef __attribute__((ext_vector_type(8)))  float    v8f;

static constexpr int SMEM_BYTES = 196608; // 192 KB (<= 320 KB/WGP on CDNA5)

// LDS byte offsets (wqkv region is dead after phase 1 and is reused for P/AO)
#define OFF_WQKV   0        // 384*128 f16 = 98304 B   (phases 0-1)
#define OFF_P      0        // 4*64*64 f16 = 32768 B   (phases 2-3, aliases wqkv)
#define OFF_AO     32768    // 64*128  f16 = 16384 B   (phases 3-4, aliases wqkv)
#define OFF_WPROJ  98304    // 128*128 f16 = 32768 B
#define OFF_X      131072   // 64*128  f16 = 16384 B
#define OFF_Q      147456   // 4*64*32 f16 = 16384 B
#define OFF_K      163840   // 4*64*32 f16 = 16384 B
#define OFF_VT     180224   // 4*32*64 f16 = 16384 B  (v stored transposed [h][d][n])

// Load one 16x(K=32) operand tile in the CDNA5 WMMA 16-bit A/B register layout:
// lane r=lane&15 -> row, half=lane>>4 -> K-offset; element e holds
// K = (e<8 ? 0 : 16) + half*8 + (e&7). Source is row-major with leading dim ld.
__device__ __forceinline__ v16h ld_tile(const _Float16* base, int ld) {
  const int lane = threadIdx.x & 31;
  const int r    = lane & 15;
  const int hf   = lane >> 4;
  const _Float16* p = base + r * ld + hf * 8;
  v16h a;
#pragma unroll
  for (int e = 0; e < 8; ++e) a[e] = p[e];          // K = hf*8 + e
#pragma unroll
  for (int e = 0; e < 8; ++e) a[e + 8] = p[16 + e]; // K = 16 + hf*8 + e
  return a;
}

__device__ __forceinline__ v8f wmma_f16(v16h a, v16h b, v8f c) {
  return __builtin_amdgcn_wmma_f32_16x16x32_f16(false, a, false, b,
                                                (short)0, c, false, false);
}

__global__ void __launch_bounds__(256, 1)
swin_msa_fused(const float* __restrict__ x, const float* __restrict__ mask,
               const float* __restrict__ w_qkv, const float* __restrict__ b_qkv,
               const float* __restrict__ w_proj, const float* __restrict__ b_proj,
               const float* __restrict__ bias_table, const int* __restrict__ rel_index,
               float* __restrict__ out) {
  extern __shared__ __align__(16) char smem[];
  _Float16* smemh  = (_Float16*)smem;
  _Float16* wqkvS  = (_Float16*)(smem + OFF_WQKV);
  _Float16* Ps     = (_Float16*)(smem + OFF_P);
  _Float16* aoS    = (_Float16*)(smem + OFF_AO);
  _Float16* wprojS = (_Float16*)(smem + OFF_WPROJ);
  _Float16* xS     = (_Float16*)(smem + OFF_X);
  _Float16* qS     = (_Float16*)(smem + OFF_Q);
  _Float16* kS     = (_Float16*)(smem + OFF_K);
  _Float16* vtS    = (_Float16*)(smem + OFF_VT);

  const int b    = blockIdx.x;
  const int tid  = threadIdx.x;
  const int lane = tid & 31;
  const int wave = tid >> 5;
  const int r    = lane & 15;
  const int hf   = lane >> 4;

  // ---------------- Phase 0: stage x and weights as f16 in LDS -------------
  {
    const float4* wq4 = (const float4*)w_qkv;
    for (int i = tid; i < (384 * CDIM) / 4; i += 256) {
      float4 v = wq4[i];
      v4h h4 = {(_Float16)v.x, (_Float16)v.y, (_Float16)v.z, (_Float16)v.w};
      *(v4h*)(wqkvS + i * 4) = h4;
    }
    const float4* wp4 = (const float4*)w_proj;
    for (int i = tid; i < (CDIM * CDIM) / 4; i += 256) {
      float4 v = wp4[i];
      v4h h4 = {(_Float16)v.x, (_Float16)v.y, (_Float16)v.z, (_Float16)v.w};
      *(v4h*)(wprojS + i * 4) = h4;
    }
    const float4* x4 = (const float4*)(x + (size_t)b * NTOK * CDIM);
    for (int i = tid; i < (NPAD * CDIM) / 4; i += 256) {
      v4h h4 = {(_Float16)0.f, (_Float16)0.f, (_Float16)0.f, (_Float16)0.f};
      if ((i >> 5) < NTOK) { // row = (i*4)/128
        float4 v = x4[i];
        h4 = {(_Float16)v.x, (_Float16)v.y, (_Float16)v.z, (_Float16)v.w};
      }
      *(v4h*)(xS + i * 4) = h4;
    }
  }
  __syncthreads();

  // ---------------- Phase 1: QKV GEMM (64x384 = x(64x128) @ wqkv^T) --------
  // nt-outer: each wave owns 3 output-channel tiles (nt = wave*3 + j) and all
  // 4 token tiles. B tiles (wqkv) loaded once per nt, reused for 4 mt.
  const float scale = 0.17677669529663687f; // D^-0.5
#pragma unroll 1
  for (int j = 0; j < 3; ++j) {
    const int nt = wave * 3 + j;
    v16h bm[4];
#pragma unroll
    for (int kk = 0; kk < 4; ++kk)
      bm[kk] = ld_tile(wqkvS + nt * 16 * CDIM + kk * 32, CDIM);

    // Per-lane output-column metadata (uniform sec within a tile: 16 | 128).
    const int cg  = nt * 16 + r;       // global out channel in [0,384)
    const int sec = cg >> 7;           // 0=q 1=k 2=v
    const int hh  = (cg >> 5) & 3;
    const int dd  = cg & 31;
    const float smul = (sec == 0) ? scale : 1.0f;
    const float bqs  = b_qkv[cg] * smul;
    const int stride = (sec == 2) ? 1 : DHEAD;      // vT is token-contiguous
    int colbase;
    if (sec == 0)      colbase = OFF_Q  / 2 + hh * 2048 + dd;
    else if (sec == 1) colbase = OFF_K  / 2 + hh * 2048 + dd;
    else               colbase = OFF_VT / 2 + hh * 2048 + dd * NPAD;

#pragma unroll 1
    for (int mt = 0; mt < 4; ++mt) {
      v8f acc = {};
#pragma unroll
      for (int kk = 0; kk < 4; ++kk)
        acc = wmma_f16(ld_tile(xS + mt * 16 * CDIM + kk * 32, CDIM), bm[kk], acc);
      const int a0 = colbase + (mt * 16 + hf * 8) * stride;
#pragma unroll
      for (int i = 0; i < 8; ++i)
        smemh[a0 + i * stride] = (_Float16)fmaf(acc[i], smul, bqs);
    }
  }
  __syncthreads();

  // ---------------- Phase 2: S = q @ k^T, +bias +mask, softmax -> P --------
  // 2 waves per head; wave part p owns score rows [p*32, p*32+32).
  const int h = wave >> 1;
  const int p = wave & 1;
  v8f sacc[2][4];
  {
    v16h bk[4];
#pragma unroll
    for (int tc = 0; tc < 4; ++tc)
      bk[tc] = ld_tile(kS + h * 2048 + tc * 16 * DHEAD, DHEAD);
#pragma unroll
    for (int trl = 0; trl < 2; ++trl) {
      v16h aq = ld_tile(qS + h * 2048 + (p * 2 + trl) * 16 * DHEAD, DHEAD);
#pragma unroll
      for (int tc = 0; tc < 4; ++tc) {
        v8f z = {};
        sacc[trl][tc] = wmma_f16(aq, bk[tc], z); // K = 32 in one step
      }
    }
  }
  const float* maskw = mask + (size_t)(b & 63) * NTOK * NTOK;
#pragma unroll
  for (int trl = 0; trl < 2; ++trl) {
#pragma unroll
    for (int i = 0; i < 8; ++i) {
      const int m = (p * 2 + trl) * 16 + hf * 8 + i;
      float row[4];
#pragma unroll
      for (int tc = 0; tc < 4; ++tc) {
        const int n = tc * 16 + r;
        float v = sacc[trl][tc][i];
        if (m < NTOK && n < NTOK) {
          const int ri = rel_index[m * NTOK + n];
          v += bias_table[ri * HEADS + h] + maskw[m * NTOK + n];
        } else {
          v = -1e30f; // mask pad columns
        }
        row[tc] = v;
      }
      // row max over 4 regs x 16 lanes (xor 1..8 stays inside the half)
      float mx = fmaxf(fmaxf(row[0], row[1]), fmaxf(row[2], row[3]));
#pragma unroll
      for (int off = 1; off < 16; off <<= 1) mx = fmaxf(mx, __shfl_xor(mx, off, 32));
      const float e0 = __expf(row[0] - mx), e1 = __expf(row[1] - mx);
      const float e2 = __expf(row[2] - mx), e3 = __expf(row[3] - mx);
      float sum = (e0 + e1) + (e2 + e3);
#pragma unroll
      for (int off = 1; off < 16; off <<= 1) sum += __shfl_xor(sum, off, 32);
      const float inv = 1.0f / sum;
      _Float16* prow = Ps + h * 4096 + m * NPAD;
      prow[0 * 16 + r] = (_Float16)(e0 * inv);
      prow[1 * 16 + r] = (_Float16)(e1 * inv);
      prow[2 * 16 + r] = (_Float16)(e2 * inv);
      prow[3 * 16 + r] = (_Float16)(e3 * inv);
    }
  }
  __syncthreads();

  // ---------------- Phase 3: out_h = P(64x64) @ v(64x32), K = 2 steps ------
  {
    v8f oacc[2][2];
    v8f z = {};
    oacc[0][0] = z; oacc[0][1] = z; oacc[1][0] = z; oacc[1][1] = z;
#pragma unroll
    for (int kk = 0; kk < 2; ++kk) {
      v16h bv0 = ld_tile(vtS + h * 2048 + 0  * NPAD + kk * 32, NPAD);
      v16h bv1 = ld_tile(vtS + h * 2048 + 16 * NPAD + kk * 32, NPAD);
      v16h a0  = ld_tile(Ps + h * 4096 + (p * 2 + 0) * 16 * NPAD + kk * 32, NPAD);
      v16h a1  = ld_tile(Ps + h * 4096 + (p * 2 + 1) * 16 * NPAD + kk * 32, NPAD);
      oacc[0][0] = wmma_f16(a0, bv0, oacc[0][0]);
      oacc[0][1] = wmma_f16(a0, bv1, oacc[0][1]);
      oacc[1][0] = wmma_f16(a1, bv0, oacc[1][0]);
      oacc[1][1] = wmma_f16(a1, bv1, oacc[1][1]);
    }
#pragma unroll
    for (int trl = 0; trl < 2; ++trl)
#pragma unroll
      for (int tc = 0; tc < 2; ++tc) {
        const int m0 = (p * 2 + trl) * 16 + hf * 8;
        const int dd = tc * 16 + r;
#pragma unroll
        for (int i = 0; i < 8; ++i)
          aoS[(m0 + i) * CDIM + h * DHEAD + dd] = (_Float16)oacc[trl][tc][i];
      }
  }
  __syncthreads();

  // ---------------- Phase 4: proj = ao(64x128) @ wproj^T + b_proj ----------
  // mt-outer: A tiles (attn-out rows) loaded once per wave, reused for 4 nt.
  {
    const int mt = wave >> 1;
    const int nbase = (wave & 1) * 4;
    const size_t outb = (size_t)b * NTOK * CDIM;
    v16h am[4];
#pragma unroll
    for (int kk = 0; kk < 4; ++kk)
      am[kk] = ld_tile(aoS + mt * 16 * CDIM + kk * 32, CDIM);
#pragma unroll 1
    for (int j = 0; j < 4; ++j) {
      const int nt = nbase + j;
      v8f acc = {};
#pragma unroll
      for (int kk = 0; kk < 4; ++kk)
        acc = wmma_f16(am[kk], ld_tile(wprojS + nt * 16 * CDIM + kk * 32, CDIM), acc);
      const int n = nt * 16 + r;
      const float bp = b_proj[n];
#pragma unroll
      for (int i = 0; i < 8; ++i) {
        const int m = mt * 16 + hf * 8 + i;
        if (m < NTOK) out[outb + m * CDIM + n] = acc[i] + bp;
      }
    }
  }
}

extern "C" void kernel_launch(void* const* d_in, const int* in_sizes, int n_in,
                              void* d_out, int out_size, void* d_ws, size_t ws_size,
                              hipStream_t stream) {
  (void)n_in; (void)d_ws; (void)ws_size; (void)out_size;
  const float* x          = (const float*)d_in[0];
  const float* mask       = (const float*)d_in[1];
  const float* w_qkv      = (const float*)d_in[2];
  const float* b_qkv      = (const float*)d_in[3];
  const float* w_proj     = (const float*)d_in[4];
  const float* b_proj     = (const float*)d_in[5];
  const float* bias_table = (const float*)d_in[6];
  const int*   rel_index  = (const int*)d_in[7];
  float* out = (float*)d_out;

  const int B = in_sizes[0] / (NTOK * CDIM); // 4096 windows

  // Opt in to >default dynamic LDS (192 KB of the 320 KB/WGP on CDNA5).
  (void)hipFuncSetAttribute(reinterpret_cast<const void*>(swin_msa_fused),
                            hipFuncAttributeMaxDynamicSharedMemorySize, SMEM_BYTES);

  swin_msa_fused<<<B, 256, SMEM_BYTES, stream>>>(
      x, mask, w_qkv, b_qkv, w_proj, b_proj, bias_table, rel_index, out);
}
